// BidirectionalMamba_18717467476145
// MI455X (gfx1250) — matrix operand order repaired and checked by reading, not compiled, not run on hardware
//
#include <hip/hip_runtime.h>
#include <math.h>

// ---------------------------------------------------------------------------
// Bidirectional Mamba for MI455X (gfx1250, wave32, WMMA).
// GEMMs: C[M x N] = A[M x K](bf16) @ W[N x K]^T(bf16), f32 acc via
// v_wmma_f32_16x16x32_bf16.  All K padded to multiples of 32 so every
// fragment load is an unconditional, aligned, contiguous vector load.
// Selective scan stays fp32 in registers, B/C staged in LDS.
// ---------------------------------------------------------------------------

#define DMODEL 768
#define DINNER 1536
#define DTRANK 48
#define DTPAD  64          // DT_RANK zero-padded to a K-multiple of 32
#define DSTATE 16
#define BATCH  4
#define SEQ    1024
#define TOK    (BATCH * SEQ)   // 4096 rows, multiple of 128

typedef __attribute__((ext_vector_type(16))) __bf16 v16bf;
typedef __attribute__((ext_vector_type(8)))  __bf16 bf16x8;
typedef __attribute__((ext_vector_type(8)))  float  v8f;

enum { EPI_NONE = 0, EPI_BIAS = 1, EPI_SOFTPLUS_BIAS = 2 };

__device__ __forceinline__ float silu_f(float v) {
    return v / (1.0f + __expf(-v));
}

// ---------------------------------------------------------------------------
// fp32 -> bf16 conversion
// ---------------------------------------------------------------------------
__global__ void cvt_f32_bf16(const float* __restrict__ src,
                             __bf16* __restrict__ dst, int n) {
    int i = blockIdx.x * blockDim.x + threadIdx.x;
    if (i < n) dst[i] = (__bf16)src[i];
}

// fp32 (rows x srcK) -> bf16 (rows x dstK), zero-padded columns
__global__ void cvt_pad_f32_bf16(const float* __restrict__ src,
                                 __bf16* __restrict__ dst,
                                 int srcK, int dstK, int n /* rows*dstK */) {
    int i = blockIdx.x * blockDim.x + threadIdx.x;
    if (i < n) {
        int row = i / dstK, col = i - row * dstK;
        dst[i] = (col < srcK) ? (__bf16)src[(size_t)row * srcK + col] : (__bf16)0.0f;
    }
}

// ---------------------------------------------------------------------------
// WMMA GEMM.  Block = 256 threads = 8 waves (4 M-waves x 2 N-waves).
// Block tile 128(M) x 128(N); wave tile 32(M) x 64(N) = 8 accumulators;
// per K-step (32): 2 A-frag loads + 4 B-frag loads -> 8 WMMAs.
// Requires: K % 32 == 0, M % 128 == 0, N % 16 == 0, all lda/ldw/ldc used
// here give 32B-aligned fragment addresses.
// revMask!=0 reads A rows sequence-flipped within each batch (row ^ 1023).
// ---------------------------------------------------------------------------
template <int EPI, typename OutT>
__global__ __launch_bounds__(256)
void gemm_bf16_wmma(const __bf16* __restrict__ A, int lda,
                    const __bf16* __restrict__ W, int ldw,
                    OutT* __restrict__ C, int ldc,
                    const float* __restrict__ bias,
                    int N, int K, int revMask) {
    const int lane    = threadIdx.x & 31;
    const int wave    = threadIdx.x >> 5;
    const int m0      = blockIdx.y * 128 + (wave & 3) * 32;
    const int n0      = blockIdx.x * 128 + (wave >> 2) * 64;
    const int halfSel = (lane >> 4) & 1;   // 0: lanes 0-15, 1: lanes 16-31
    const int laneN   = lane & 15;

    // A fragment (16-bit A 16x32, this code's contiguous k order): lane holds row m,
    // elems 0..15 = K kb..kb+15, kb = halfSel?16:0 (matches B).
    const int kbA = halfSel ? 16 : 0;
    // B fragment (32x16): lane holds col n, elems 0..15 = K kb..kb+15, kb = halfSel?16:0.
    const int kbB = halfSel ? 16 : 0;

    const __bf16* Arow0 = A + (size_t)((m0 + laneN) ^ revMask) * lda + kbA;
    const __bf16* Arow1 = A + (size_t)((m0 + 16 + laneN) ^ revMask) * lda + kbA;

    bool nval[4];
    const __bf16* Wp[4];
#pragma unroll
    for (int j = 0; j < 4; ++j) {
        int nbase = n0 + 16 * j;
        nval[j] = (nbase < N);                 // N is a multiple of 16
        int nb = nval[j] ? nbase : 0;          // clamp for compute; store guarded
        Wp[j] = W + (size_t)(nb + laneN) * ldw + kbB;
    }

    v8f acc[8];
#pragma unroll
    for (int q = 0; q < 8; ++q)
#pragma unroll
        for (int r = 0; r < 8; ++r) acc[q][r] = 0.0f;

    union AFrag { v16bf v; bf16x8 h[2]; };

    for (int k = 0; k < K; k += 32) {
        AFrag a0, a1;
        a0.h[0] = *reinterpret_cast<const bf16x8*>(Arow0 + k);
        a0.h[1] = *reinterpret_cast<const bf16x8*>(Arow0 + k + 8);
        a1.h[0] = *reinterpret_cast<const bf16x8*>(Arow1 + k);
        a1.h[1] = *reinterpret_cast<const bf16x8*>(Arow1 + k + 8);
#pragma unroll
        for (int j = 0; j < 4; ++j) {
            v16bf bfrag = *reinterpret_cast<const v16bf*>(Wp[j] + k);
            acc[j] = __builtin_amdgcn_wmma_f32_16x16x32_bf16(
                false, a0.v, false, bfrag, (short)0, acc[j], false, false);
            acc[4 + j] = __builtin_amdgcn_wmma_f32_16x16x32_bf16(
                false, a1.v, false, bfrag, (short)0, acc[4 + j], false, false);
        }
    }

    // C/D layout: VGPR r -> row base+r (lanes 0-15) / base+8+r (lanes 16-31),
    // col = n + lane%16.
    const int rBase = halfSel ? 8 : 0;
#pragma unroll
    for (int j = 0; j < 4; ++j) {
        if (!nval[j]) continue;                // wave-uniform branch
        int n = n0 + 16 * j + laneN;
        float bv = (EPI != EPI_NONE) ? bias[n] : 0.0f;
#pragma unroll
        for (int mi = 0; mi < 2; ++mi) {
#pragma unroll
            for (int r = 0; r < 8; ++r) {
                int row = m0 + mi * 16 + rBase + r;
                float v = acc[mi * 4 + j][r];
                if (EPI == EPI_BIAS) v += bv;
                if (EPI == EPI_SOFTPLUS_BIAS) {
                    v += bv;
                    v = (v > 20.0f) ? v : log1pf(__expf(v));
                }
                C[(size_t)row * ldc + n] = (OutT)v;
            }
        }
    }
}

// ---------------------------------------------------------------------------
// Depthwise causal conv (D_CONV=4) + bias + SiLU over xi half of xz.
// grid = (DINNER/256, TOK)
// ---------------------------------------------------------------------------
__global__ __launch_bounds__(256)
void conv_silu_kernel(const __bf16* __restrict__ xz,
                      const float* __restrict__ cw,   // (DINNER,4)
                      const float* __restrict__ cb,   // (DINNER)
                      float* __restrict__ xcf,
                      __bf16* __restrict__ xcb) {
    int c = blockIdx.x * 256 + threadIdx.x;   // channel
    int t = blockIdx.y;                       // token (b*SEQ + l)
    int l = t & (SEQ - 1);
    float acc = cb[c];
#pragma unroll
    for (int k = 0; k < 4; ++k) {
        int ls = l - 3 + k;
        if (ls >= 0)
            acc += cw[c * 4 + k] * (float)xz[(size_t)(t - 3 + k) * (2 * DINNER) + c];
    }
    float v = silu_f(acc);
    xcf[(size_t)t * DINNER + c] = v;
    xcb[(size_t)t * DINNER + c] = (__bf16)v;
}

// ---------------------------------------------------------------------------
// Extract dt (cols 0..47 of x_dbl) to bf16, zero-padded to DTPAD columns.
// ---------------------------------------------------------------------------
__global__ void extract_dt(const float* __restrict__ xdbl,
                           __bf16* __restrict__ dtb, int n /* TOK*DTPAD */) {
    int i = blockIdx.x * blockDim.x + threadIdx.x;
    if (i < n) {
        int t = i / DTPAD, r = i - t * DTPAD;
        dtb[i] = (r < DTRANK) ? (__bf16)xdbl[(size_t)t * 80 + r] : (__bf16)0.0f;
    }
}

// ---------------------------------------------------------------------------
// Selective scan + skip (D) + SiLU(z) gate, output bf16.
// grid = (DINNER/256, BATCH).  Per thread: one channel, h[16]+A[16] in VGPRs.
// B_t/C_t (32 floats/step) staged in LDS by the first 32 lanes.
// ---------------------------------------------------------------------------
__global__ __launch_bounds__(256)
void scan_kernel(const float* __restrict__ delta,   // (TOK, DINNER)
                 const float* __restrict__ xcf,     // (TOK, DINNER)  u
                 const float* __restrict__ xdbl,    // (TOK, 80)  B@48, C@64
                 const __bf16* __restrict__ xz,     // (TOK, 2*DINNER)  z at +DINNER
                 const float* __restrict__ Alog,    // (DINNER, 16)
                 const float* __restrict__ Dp,      // (DINNER)
                 __bf16* __restrict__ ybf) {        // (TOK, DINNER)
    __shared__ float sBC[32];
    const int c = blockIdx.x * 256 + threadIdx.x;
    const int b = blockIdx.y;

    float Arow[DSTATE];
#pragma unroll
    for (int s = 0; s < DSTATE; ++s) Arow[s] = -__expf(Alog[(size_t)c * DSTATE + s]);
    const float Dv = Dp[c];

    float h[DSTATE];
#pragma unroll
    for (int s = 0; s < DSTATE; ++s) h[s] = 0.0f;

    for (int l = 0; l < SEQ; ++l) {
        const int t = b * SEQ + l;
        __syncthreads();  // previous step's readers done
        if (threadIdx.x < 32) sBC[threadIdx.x] = xdbl[(size_t)t * 80 + 48 + threadIdx.x];
        __syncthreads();

        const float d  = delta[(size_t)t * DINNER + c];
        const float u  = xcf[(size_t)t * DINNER + c];
        const float du = d * u;
        float y = 0.0f;
#pragma unroll
        for (int s = 0; s < DSTATE; ++s) {
            h[s] = __expf(d * Arow[s]) * h[s] + du * sBC[s];
            y += h[s] * sBC[16 + s];
        }
        y += u * Dv;
        const float z = (float)xz[(size_t)t * (2 * DINNER) + DINNER + c];
        y *= silu_f(z);
        ybf[(size_t)t * DINNER + c] = (__bf16)y;
    }
}

// ---------------------------------------------------------------------------
// Host launcher
// ---------------------------------------------------------------------------
extern "C" void kernel_launch(void* const* d_in, const int* in_sizes, int n_in,
                              void* d_out, int out_size, void* d_ws, size_t ws_size,
                              hipStream_t stream) {
    (void)in_sizes; (void)n_in; (void)out_size; (void)ws_size;

    const float* x        = (const float*)d_in[0];
    const float* inW[2]   = {(const float*)d_in[1],  (const float*)d_in[10]};
    const float* convw[2] = {(const float*)d_in[2],  (const float*)d_in[11]};
    const float* convb[2] = {(const float*)d_in[3],  (const float*)d_in[12]};
    const float* xprW[2]  = {(const float*)d_in[4],  (const float*)d_in[13]};
    const float* dtW[2]   = {(const float*)d_in[5],  (const float*)d_in[14]};
    const float* dtb[2]   = {(const float*)d_in[6],  (const float*)d_in[15]};
    const float* Alog[2]  = {(const float*)d_in[7],  (const float*)d_in[16]};
    const float* Dp[2]    = {(const float*)d_in[8],  (const float*)d_in[17]};
    const float* outW[2]  = {(const float*)d_in[9],  (const float*)d_in[18]};
    const float* projW    = (const float*)d_in[19];
    const float* projb    = (const float*)d_in[20];
    float*       out      = (float*)d_out;

    // -------- workspace carve (256B aligned) --------
    char* wp = (char*)d_ws;
    auto alloc = [&](size_t elems, size_t esz) -> void* {
        void* r = (void*)wp;
        wp += (elems * esz + 255) & ~(size_t)255;
        return r;
    };
    __bf16* x_bf      = (__bf16*)alloc((size_t)TOK * DMODEL, 2);
    __bf16* inWbf[2]  = {(__bf16*)alloc((size_t)2 * DINNER * DMODEL, 2),
                         (__bf16*)alloc((size_t)2 * DINNER * DMODEL, 2)};
    __bf16* xprWbf[2] = {(__bf16*)alloc((size_t)80 * DINNER, 2),
                         (__bf16*)alloc((size_t)80 * DINNER, 2)};
    __bf16* dtWbf[2]  = {(__bf16*)alloc((size_t)DINNER * DTPAD, 2),   // zero-padded K
                         (__bf16*)alloc((size_t)DINNER * DTPAD, 2)};
    __bf16* outWbf[2] = {(__bf16*)alloc((size_t)DMODEL * DINNER, 2),
                         (__bf16*)alloc((size_t)DMODEL * DINNER, 2)};
    __bf16* projWbf   = (__bf16*)alloc((size_t)DMODEL * 2 * DMODEL, 2); // 768x1536
    __bf16* xz_bf     = (__bf16*)alloc((size_t)TOK * 2 * DINNER, 2);
    float*  xc_f      = (float*) alloc((size_t)TOK * DINNER, 4);
    __bf16* xc_bf     = (__bf16*)alloc((size_t)TOK * DINNER, 2);
    float*  xdbl      = (float*) alloc((size_t)TOK * 80, 4);
    __bf16* dt_bf     = (__bf16*)alloc((size_t)TOK * DTPAD, 2);        // zero-padded K
    float*  delta     = (float*) alloc((size_t)TOK * DINNER, 4);
    __bf16* y_bf      = (__bf16*)alloc((size_t)TOK * DINNER, 2);
    __bf16* ycat      = (__bf16*)alloc((size_t)TOK * DINNER, 2);

    auto cvt = [&](const float* s, __bf16* d, int n) {
        cvt_f32_bf16<<<(n + 255) / 256, 256, 0, stream>>>(s, d, n);
    };

    // -------- per-launch bf16 conversions --------
    cvt(x, x_bf, TOK * DMODEL);
    for (int dir = 0; dir < 2; ++dir) {
        cvt(inW[dir],  inWbf[dir],  2 * DINNER * DMODEL);
        cvt(xprW[dir], xprWbf[dir], 80 * DINNER);
        cvt_pad_f32_bf16<<<(DINNER * DTPAD + 255) / 256, 256, 0, stream>>>(
            dtW[dir], dtWbf[dir], DTRANK, DTPAD, DINNER * DTPAD);
        cvt(outW[dir], outWbf[dir], DMODEL * DINNER);
    }
    cvt(projW, projWbf, DMODEL * 2 * DMODEL);

    const dim3 blk(256);
    const int gy = TOK / 128;  // 32

    for (int dir = 0; dir < 2; ++dir) {
        const int rev = dir ? (SEQ - 1) : 0;

        // in-proj: (TOK x 3072) = x(rev?) @ inW^T, K=768
        gemm_bf16_wmma<EPI_NONE, __bf16>
            <<<dim3(2 * DINNER / 128, gy), blk, 0, stream>>>(
                x_bf, DMODEL, inWbf[dir], DMODEL, xz_bf, 2 * DINNER,
                nullptr, 2 * DINNER, DMODEL, rev);

        // depthwise conv + SiLU
        conv_silu_kernel<<<dim3(DINNER / 256, TOK), blk, 0, stream>>>(
            xz_bf, convw[dir], convb[dir], xc_f, xc_bf);

        // x-proj: (TOK x 80), K=1536
        gemm_bf16_wmma<EPI_NONE, float>
            <<<dim3(1, gy), blk, 0, stream>>>(
                xc_bf, DINNER, xprWbf[dir], DINNER, xdbl, 80,
                nullptr, 80, DINNER, 0);

        // dt slice -> bf16 (zero-padded to DTPAD)
        extract_dt<<<(TOK * DTPAD + 255) / 256, blk, 0, stream>>>(
            xdbl, dt_bf, TOK * DTPAD);

        // dt-proj + softplus(.+dt_b): (TOK x 1536), K=DTPAD
        gemm_bf16_wmma<EPI_SOFTPLUS_BIAS, float>
            <<<dim3(DINNER / 128, gy), blk, 0, stream>>>(
                dt_bf, DTPAD, dtWbf[dir], DTPAD, delta, DINNER,
                dtb[dir], DINNER, DTPAD, 0);

        // selective scan + D skip + SiLU(z) gate
        scan_kernel<<<dim3(DINNER / 256, BATCH), blk, 0, stream>>>(
            delta, xc_f, xdbl, xz_bf, Alog[dir], Dp[dir], y_bf);

        // out-proj into concat buffer (bwd: un-flip via rev read of y), K=1536
        gemm_bf16_wmma<EPI_NONE, __bf16>
            <<<dim3(DMODEL / 128, gy), blk, 0, stream>>>(
                y_bf, DINNER, outWbf[dir], DINNER,
                ycat + (dir ? DMODEL : 0), 2 * DMODEL,
                nullptr, DMODEL, DINNER, rev);
    }

    // final projection + bias -> d_out (fp32), K=1536
    gemm_bf16_wmma<EPI_BIAS, float>
        <<<dim3(DMODEL / 128, gy), blk, 0, stream>>>(
            ycat, 2 * DMODEL, projWbf, 2 * DMODEL, out, DMODEL,
            projb, DMODEL, 2 * DMODEL, 0);
}
